// SelfAttention_69028714381895
// MI455X (gfx1250) — matrix operand order repaired and checked by reading, not compiled, not run on hardware
//
#include <hip/hip_runtime.h>

#define EMBED 1024
#define HEADS 16
#define HDIM  64

typedef __attribute__((ext_vector_type(16))) __bf16 bf16x16;
typedef __attribute__((ext_vector_type(8)))  __bf16 bf16x8;
typedef __attribute__((ext_vector_type(4)))  __bf16 bf16x4;
typedef __attribute__((ext_vector_type(8)))  float  f32x8;
typedef int v4i __attribute__((vector_size(16)));

#if __has_builtin(__builtin_amdgcn_global_load_async_to_lds_b128)
#define ASYNC_LDS 1
#else
#define ASYNC_LDS 0
#endif

// copy 16B global -> LDS (async on gfx1250 when available)
__device__ __forceinline__ void cp16_to_lds(const __bf16* gsrc, __bf16* ldst) {
#if ASYNC_LDS
  __builtin_amdgcn_global_load_async_to_lds_b128((v4i*)gsrc, (v4i*)ldst, 0, 0);
#else
  *(bf16x8*)ldst = *(const bf16x8*)gsrc;
#endif
}

__device__ __forceinline__ void wait_async_all() {
#if ASYNC_LDS
#if __has_builtin(__builtin_amdgcn_s_wait_asynccnt)
  __builtin_amdgcn_s_wait_asynccnt(0);
#else
  asm volatile("s_wait_asynccnt 0x0" ::: "memory");
#endif
#endif
}

__device__ __forceinline__ f32x8 wmma_bf16(bf16x16 a, bf16x16 b, f32x8 c) {
  return __builtin_amdgcn_wmma_f32_16x16x32_bf16(false, a, false, b, (short)0, c, false, false);
}

// load a 32B B-fragment as two 16B chunks (LDS-alignment safe)
__device__ __forceinline__ bf16x16 ld_frag32(const __bf16* p) {
  bf16x8 lo = *(const bf16x8*)p;
  bf16x8 hi = *(const bf16x8*)(p + 16);
  bf16x16 r;
#pragma unroll
  for (int i = 0; i < 8; i++) { r[i] = lo[i]; r[i + 8] = hi[i]; }
  return r;
}

// ---------------------------------------------------------------------------
// Kernel 0: pre-convert Wo f32 -> bf16 (removes cvt + halves B traffic in the
// big output GEMM; Wo_bf16 = 2MB, L2-resident for the whole GEMM)
// ---------------------------------------------------------------------------
__global__ __launch_bounds__(256) void cvt_wo_kernel(const float* __restrict__ Wo,
                                                     unsigned short* __restrict__ wo_u)
{
  size_t i = ((size_t)blockIdx.x * 256 + threadIdx.x) * 4;
  float4 f = *(const float4*)(Wo + i);
  bf16x4 o;
  o[0] = (__bf16)f.x; o[1] = (__bf16)f.y; o[2] = (__bf16)f.z; o[3] = (__bf16)f.w;
  *(bf16x4*)((__bf16*)wo_u + i) = o;
}

// ---------------------------------------------------------------------------
// Kernel 1: per-head QKV projection.  out[s,e] = sum_d X[s, h*64+d] * W[e,d]
// Q,K stored bf16 [n,h,s,d]; V stored transposed bf16 [n,h,d,s].
// ---------------------------------------------------------------------------
__global__ __launch_bounds__(128) void proj_kernel(
    const float* __restrict__ q, const float* __restrict__ k, const float* __restrict__ v,
    const float* __restrict__ Wq, const float* __restrict__ Wk, const float* __restrict__ Wv,
    unsigned short* __restrict__ qh_u, unsigned short* __restrict__ kh_u,
    unsigned short* __restrict__ vt_u, int S)
{
  const int w    = threadIdx.x >> 5;
  const int lane = threadIdx.x & 31;
  const int ln   = lane & 15, khf = lane >> 4;
  const int which = blockIdx.z % 3;
  const int n     = blockIdx.z / 3;
  const int h     = blockIdx.y;
  const int m_base = blockIdx.x * 64 + w * 16;

  const float* X = (which == 0) ? q : (which == 1) ? k : v;
  const float* W = (which == 0) ? Wq : (which == 1) ? Wk : Wv;
  X += (size_t)n * S * EMBED;

  f32x8 c[4] = {};
#pragma unroll
  for (int ks = 0; ks < 2; ks++) {
    bf16x16 a;
    const float* pa = X + (size_t)(m_base + ln) * EMBED + h * HDIM + ks * 32 + 8 * khf;
#pragma unroll
    for (int i = 0; i < 8; i++) { a[i] = (__bf16)pa[i]; a[i + 8] = (__bf16)pa[16 + i]; }
#pragma unroll
    for (int nt = 0; nt < 4; nt++) {
      bf16x16 b;
      const float* pb = W + (nt * 16 + ln) * HDIM + ks * 32 + 8 * khf;
#pragma unroll
      for (int i = 0; i < 8; i++) { b[i] = (__bf16)pb[i]; b[i + 8] = (__bf16)pb[16 + i]; }
      c[nt] = wmma_bf16(a, b, c[nt]);
    }
  }

  if (which == 2) {
    __bf16* vt = (__bf16*)vt_u + (size_t)(n * HEADS + h) * HDIM * S;
#pragma unroll
    for (int nt = 0; nt < 4; nt++) {
      bf16x8 pk;
#pragma unroll
      for (int r = 0; r < 8; r++) pk[r] = (__bf16)c[nt][r];
      *(bf16x8*)(vt + (size_t)(nt * 16 + ln) * S + m_base + 8 * khf) = pk;
    }
  } else {
    __bf16* dst = (__bf16*)((which == 0) ? qh_u : kh_u) + (size_t)(n * HEADS + h) * S * HDIM;
#pragma unroll
    for (int nt = 0; nt < 4; nt++)
#pragma unroll
      for (int r = 0; r < 8; r++)
        dst[(size_t)(m_base + 8 * khf + r) * HDIM + nt * 16 + ln] = (__bf16)c[nt][r];
  }
}

// ---------------------------------------------------------------------------
// Kernel 2: fused flash attention per (n,h).  Block = 4 waves = 64 q rows.
// K/V tiles (64x64) staged ONCE per block in LDS via async global->LDS copies
// (double-buffered: tile j+1 loads overlap compute on tile j), shared by all
// 4 waves (4x VMEM traffic reduction).  Online softmax in f32; P bounced
// through per-wave LDS to convert C-layout -> A-layout for the P@V WMMAs.
// ---------------------------------------------------------------------------
__global__ __launch_bounds__(128) void attn_kernel(
    const unsigned short* __restrict__ qh_u, const unsigned short* __restrict__ kh_u,
    const unsigned short* __restrict__ vt_u, unsigned short* __restrict__ xa_u, int S)
{
  __shared__ __attribute__((aligned(16))) __bf16 ktile[2][64][72];
  __shared__ __attribute__((aligned(16))) __bf16 vtile[2][64][72];
  __shared__ __attribute__((aligned(16))) __bf16 ptile[4][16][72];

  const int tid  = threadIdx.x;
  const int w    = tid >> 5;
  const int lane = tid & 31;
  const int ln   = lane & 15, khf = lane >> 4;
  const int n = blockIdx.z, h = blockIdx.y;
  const int qbase = blockIdx.x * 64 + w * 16;

  const __bf16* qh = (const __bf16*)qh_u + (size_t)(n * HEADS + h) * S * HDIM;
  const __bf16* kh = (const __bf16*)kh_u + (size_t)(n * HEADS + h) * S * HDIM;
  const __bf16* vt = (const __bf16*)vt_u + (size_t)(n * HEADS + h) * HDIM * S;
  __bf16* xa = (__bf16*)xa_u + (size_t)n * S * EMBED + h * HDIM;

  // stage one 64x64 K tile + one 64x64 V tile: 512 chunks of 16B each tensor
  auto issue_tiles = [&](int j, int buf) {
#pragma unroll
    for (int i = 0; i < 4; i++) {
      int c = tid + i * 128;            // 0..511
      int row = c >> 3, off = (c & 7) * 8;
      cp16_to_lds(kh + (size_t)(j * 64 + row) * HDIM + off, &ktile[buf][row][off]);
      cp16_to_lds(vt + (size_t)row * S + j * 64 + off, &vtile[buf][row][off]);
    }
  };

  // Q fragments resident for the whole pass (K-dim 64 -> 2 frags)
  bf16x16 aq[2];
#pragma unroll
  for (int ks = 0; ks < 2; ks++) {
    const __bf16* p = qh + (size_t)(qbase + ln) * HDIM + ks * 32 + 8 * khf;
    bf16x8 c0 = *(const bf16x8*)p;
    bf16x8 c1 = *(const bf16x8*)(p + 16);
#pragma unroll
    for (int i = 0; i < 8; i++) { aq[ks][i] = c0[i]; aq[ks][i + 8] = c1[i]; }
  }

  float m_i[8], l_i[8];
#pragma unroll
  for (int r = 0; r < 8; r++) { m_i[r] = -1e30f; l_i[r] = 0.0f; }
  f32x8 acc[4] = {};
  const float sc = 0.125f;  // 1/sqrt(64)
  const int NT = S / 64;

#if ASYNC_LDS
  issue_tiles(0, 0);
  wait_async_all();
  __syncthreads();
#endif

  for (int j = 0; j < NT; j++) {
#if ASYNC_LDS
    const int buf = j & 1;
    if (j + 1 < NT) issue_tiles(j + 1, buf ^ 1);   // prefetch next tile
#else
    const int buf = 0;
    __syncthreads();           // WAR: prior iteration done reading LDS
    issue_tiles(j, 0);
    __syncthreads();           // tiles visible
#endif

    // S-tile = Q @ K^T  (4 n-tiles of 16 cols, 2 k-steps of 32)
    f32x8 sf[4];
#pragma unroll
    for (int nt = 0; nt < 4; nt++) {
      f32x8 cc = {};
#pragma unroll
      for (int ks = 0; ks < 2; ks++) {
        bf16x16 bk = ld_frag32(&ktile[buf][nt * 16 + ln][ks * 32 + 8 * khf]);
        cc = wmma_bf16(aq[ks], bk, cc);
      }
      sf[nt] = cc;
    }

    // online softmax: per-row max over 64 cols (4 frags + 16-lane half-group)
    float alpha[8];
#pragma unroll
    for (int r = 0; r < 8; r++) {
      float t = fmaxf(fmaxf(sf[0][r], sf[1][r]), fmaxf(sf[2][r], sf[3][r])) * sc;
#pragma unroll
      for (int off = 1; off < 16; off <<= 1) t = fmaxf(t, __shfl_xor(t, off));
      float mn = fmaxf(m_i[r], t);
      alpha[r] = __expf(m_i[r] - mn);
      m_i[r] = mn;
    }

    float rs[8];
#pragma unroll
    for (int r = 0; r < 8; r++) rs[r] = 0.0f;
#pragma unroll
    for (int nt = 0; nt < 4; nt++) {
#pragma unroll
      for (int r = 0; r < 8; r++) {
        float p = __expf(sf[nt][r] * sc - m_i[r]);
        rs[r] += p;
        ptile[w][8 * khf + r][nt * 16 + ln] = (__bf16)p;  // C-layout -> LDS
      }
    }
#pragma unroll
    for (int r = 0; r < 8; r++) {
      float t = rs[r];
#pragma unroll
      for (int off = 1; off < 16; off <<= 1) t += __shfl_xor(t, off);
      l_i[r] = l_i[r] * alpha[r] + t;
    }
#pragma unroll
    for (int nt = 0; nt < 4; nt++)
#pragma unroll
      for (int r = 0; r < 8; r++) acc[nt][r] *= alpha[r];

    __syncthreads();  // cross-lane visibility of P tile

    // reload P in A-layout
    bf16x16 ap[2];
#pragma unroll
    for (int ks = 0; ks < 2; ks++) {
      const __bf16* p = &ptile[w][ln][ks * 32 + 8 * khf];
      bf16x8 c0 = *(const bf16x8*)p;
      bf16x8 c1 = *(const bf16x8*)(p + 16);
#pragma unroll
      for (int i = 0; i < 8; i++) { ap[ks][i] = c0[i]; ap[ks][i + 8] = c1[i]; }
    }

    // acc += P @ V   (V^T tile in LDS: row d, contiguous s)
#pragma unroll
    for (int nt = 0; nt < 4; nt++) {
#pragma unroll
      for (int ks = 0; ks < 2; ks++) {
        bf16x16 bv = ld_frag32(&vtile[buf][nt * 16 + ln][ks * 32 + 8 * khf]);
        acc[nt] = wmma_bf16(ap[ks], bv, acc[nt]);
      }
    }

#if ASYNC_LDS
    wait_async_all();  // next tile fully in LDS
    __syncthreads();   // also protects ptile WAR
#endif
  }

  // normalize and store attention output bf16 [n,s,e] for the final GEMM
#pragma unroll
  for (int r = 0; r < 8; r++) l_i[r] = 1.0f / l_i[r];
#pragma unroll
  for (int nt = 0; nt < 4; nt++)
#pragma unroll
    for (int r = 0; r < 8; r++)
      xa[(size_t)(qbase + 8 * khf + r) * EMBED + nt * 16 + ln] =
          (__bf16)(acc[nt][r] * l_i[r]);
}

// ---------------------------------------------------------------------------
// Kernel 3: output projection  out[s,e] = sum_c Xattn[s,c] * Wo_bf16[e,c]
// Pure-bf16 operand loads (32B/lane B-fragments), f32 accumulate/store.
// ---------------------------------------------------------------------------
__global__ __launch_bounds__(128) void outproj_kernel(
    const unsigned short* __restrict__ xa_u, const unsigned short* __restrict__ wo_u,
    float* __restrict__ out)
{
  const int w    = threadIdx.x >> 5;
  const int lane = threadIdx.x & 31;
  const int ln   = lane & 15, khf = lane >> 4;
  const int rbase = blockIdx.x * 64 + w * 16;
  const int cbase = blockIdx.y * 64;
  const __bf16* X  = (const __bf16*)xa_u;
  const __bf16* Wo = (const __bf16*)wo_u;

  f32x8 c[4] = {};
  for (int kc = 0; kc < EMBED; kc += 32) {
    bf16x16 a;
    const __bf16* pa = X + (size_t)(rbase + ln) * EMBED + kc + 8 * khf;
    bf16x8 c0 = *(const bf16x8*)pa;
    bf16x8 c1 = *(const bf16x8*)(pa + 16);
#pragma unroll
    for (int i = 0; i < 8; i++) { a[i] = c0[i]; a[i + 8] = c1[i]; }
#pragma unroll
    for (int nt = 0; nt < 4; nt++) {
      bf16x16 b = ld_frag32(Wo + (size_t)(cbase + nt * 16 + ln) * EMBED + kc + 8 * khf);
      c[nt] = wmma_bf16(a, b, c[nt]);
    }
  }
#pragma unroll
  for (int nt = 0; nt < 4; nt++)
#pragma unroll
    for (int r = 0; r < 8; r++)
      out[(size_t)(rbase + 8 * khf + r) * EMBED + cbase + nt * 16 + ln] = c[nt][r];
}

// ---------------------------------------------------------------------------
extern "C" void kernel_launch(void* const* d_in, const int* in_sizes, int n_in,
                              void* d_out, int out_size, void* d_ws, size_t ws_size,
                              hipStream_t stream)
{
  const float* k  = (const float*)d_in[0];
  const float* q  = (const float*)d_in[1];
  const float* v  = (const float*)d_in[2];
  const float* Wk = (const float*)d_in[3];
  const float* Wq = (const float*)d_in[4];
  const float* Wv = (const float*)d_in[5];
  const float* Wo = (const float*)d_in[6];

  const int S = 2048;
  const int T = in_sizes[0] / EMBED;  // N*S tokens
  const int N = T / S;

  size_t per_bytes = (size_t)T * EMBED * sizeof(unsigned short);
  char* wsb = (char*)d_ws;
  unsigned short* qh = (unsigned short*)(wsb);
  unsigned short* kh = (unsigned short*)(wsb + per_bytes);
  unsigned short* vt = (unsigned short*)(wsb + 2 * per_bytes);
  unsigned short* xa = (unsigned short*)(wsb + 3 * per_bytes);
  unsigned short* wo = (unsigned short*)(wsb + 4 * per_bytes);

  cvt_wo_kernel<<<dim3((EMBED * EMBED / 4) / 256), 256, 0, stream>>>(Wo, wo);
  proj_kernel<<<dim3(S / 64, HEADS, 3 * N), 128, 0, stream>>>(
      q, k, v, Wq, Wk, Wv, qh, kh, vt, S);
  attn_kernel<<<dim3(S / 64, HEADS, N), 128, 0, stream>>>(qh, kh, vt, xa, S);
  outproj_kernel<<<dim3(T / 64, EMBED / 64), 128, 0, stream>>>(xa, wo, (float*)d_out);
}
